// HiViT_89850715833038
// MI455X (gfx1250) — matrix operand order repaired and checked by reading, not compiled, not run on hardware
//
#include <hip/hip_runtime.h>
#include <math.h>

// ---------------------------------------------------------------------------
// Types for CDNA5 WMMA (gfx1250, wave32)
// ---------------------------------------------------------------------------
typedef __attribute__((ext_vector_type(16))) __bf16 v16bf;
typedef __attribute__((ext_vector_type(8)))  float  v8f;
typedef unsigned short u16;

struct alignas(16) U4 { unsigned int x, y, z, w; };
union Frag { U4 u[2]; v16bf v; };

__device__ __forceinline__ u16 f2bf(float f) {
    unsigned int u = __builtin_bit_cast(unsigned int, f);
    u = (u + 0x7FFFu + ((u >> 16) & 1u)) >> 16;
    return (u16)u;
}
__device__ __forceinline__ float bf2f(u16 h) {
    unsigned int u = ((unsigned int)h) << 16;
    return __builtin_bit_cast(float, u);
}
__device__ __forceinline__ float gelu_exact(float x) {
    return 0.5f * x * (1.0f + erff(x * 0.70710678118654752f));
}

// ---------------------------------------------------------------------------
// WMMA GEMM:  C[M,N] = act(A[M,K] @ Bt[N,K]^T + bias)
//   A, Bt bf16 (as u16), fp32 accumulate through v_wmma_f32_16x16x32_bf16.
//   Tile staging global->LDS uses CDNA5 async-to-LDS (ASYNCcnt) on the
//   aligned fast path; guarded scalar ds stores only on edge tiles.
//   OUT_MODE: 0 = store bf16, 1 = store f32, 2 = add into f32 (residual)
//   ACT: 0 = none, 1 = exact GELU
//   Batched: z in grid.z;  A += z*a_bs, Bt += z*b_bs,
//            C += (z/c_div)*c_bs1 + (z%c_div)*c_bs2
// ---------------------------------------------------------------------------
#define TM 128
#define TN 128
#define TK 32
#define LDP 40   // padded LDS row stride (halfs): 80B rows, bank friendly

__device__ __forceinline__ void stage_tile(const u16* __restrict__ G, int rows,
                                           int ld, int K, int r0, int k0,
                                           u16* __restrict__ S, int tid) {
    int lr = tid >> 1;           // 0..127 row inside tile
    int lc = (tid & 1) << 4;     // 0 or 16 halfs
    int gr = r0 + lr;
    u16* dst = S + lr * LDP + lc;
    int k = k0 + lc;
    const u16* src = G + (long long)gr * ld + k;
    if (gr < rows && (k + 16) <= K) {
        // CDNA5 async copy: global memory straight into LDS; bypasses VGPRs,
        // tracked by ASYNCcnt (waited before the tile barrier).
        unsigned           d0 = (unsigned)(uintptr_t)dst;   // low 32 bits = LDS offset
        unsigned long long a0 = (unsigned long long)(uintptr_t)src;
        asm volatile("global_load_async_to_lds_b128 %0, %1, off"
                     :: "v"(d0), "v"(a0) : "memory");
        asm volatile("global_load_async_to_lds_b128 %0, %1, off offset:16"
                     :: "v"(d0), "v"(a0) : "memory");
    } else {
        for (int i = 0; i < 16; ++i) {
            u16 v = 0;
            if (gr < rows && (k + i) < K) v = src[i];
            dst[i] = v;
        }
    }
}

__device__ __forceinline__ void wait_async_tiles() {
    // All of this wave's async-to-LDS copies complete before the barrier.
    asm volatile("s_wait_asynccnt 0" ::: "memory");
}

template <int OUT_MODE, int ACT>
__global__ __launch_bounds__(256) void gemm_bf16(
    const u16* __restrict__ A, const u16* __restrict__ Bt,
    const float* __restrict__ bias, void* __restrict__ Cv,
    int M, int N, int K, int lda, int ldb, int ldc,
    long long a_bs, long long b_bs, int c_div, long long c_bs1, long long c_bs2)
{
    __shared__ u16 As[2 * TM * LDP];
    __shared__ u16 Bs[2 * TN * LDP];

    int z = blockIdx.z;
    A  += (long long)z * a_bs;
    Bt += (long long)z * b_bs;
    long long coff = (long long)(z / c_div) * c_bs1 + (long long)(z % c_div) * c_bs2;

    int tid  = threadIdx.x;
    int lane = tid & 31;
    int wave = tid >> 5;
    int wm   = wave & 3;   // 4 wave-rows  -> 128 M
    int wn   = wave >> 2;  // 2 wave-cols  -> 128 N

    int m0 = blockIdx.y * TM;
    int n0 = blockIdx.x * TN;

    v8f acc[2][4];
    for (int i = 0; i < 2; ++i)
        for (int j = 0; j < 4; ++j)
            for (int e = 0; e < 8; ++e) acc[i][j][e] = 0.0f;

    int nk = (K + TK - 1) / TK;
    stage_tile(A,  M, lda, K, m0, 0, As, tid);
    stage_tile(Bt, N, ldb, K, n0, 0, Bs, tid);
    wait_async_tiles();
    __syncthreads();

    int rsel = lane & 15;
    int aklo = (lane & 16) ? 8 : 0;  // A frag: k = (e&8)*2 + aklo + (e&7)
    int bklo = (lane & 16) ? 8 : 0;  // B frag: k = (e&8)*2 + bklo + (e&7)

    for (int kt = 0; kt < nk; ++kt) {
        int buf = kt & 1;
        if (kt + 1 < nk) {
            int nb = (kt + 1) & 1;
            stage_tile(A,  M, lda, K, m0, (kt + 1) * TK, As + nb * TM * LDP, tid);
            stage_tile(Bt, N, ldb, K, n0, (kt + 1) * TK, Bs + nb * TN * LDP, tid);
        }
        Frag af[2];
        for (int mi = 0; mi < 2; ++mi) {
            const u16* p = As + buf * TM * LDP + (wm * 32 + mi * 16 + rsel) * LDP;
            af[mi].u[0] = *(const U4*)(p + aklo);
            af[mi].u[1] = *(const U4*)(p + aklo + 16);
        }
        Frag bfr[4];
        for (int ni = 0; ni < 4; ++ni) {
            const u16* p = Bs + buf * TN * LDP + (wn * 64 + ni * 16 + rsel) * LDP;
            bfr[ni].u[0] = *(const U4*)(p + bklo);
            bfr[ni].u[1] = *(const U4*)(p + bklo + 16);
        }
        for (int mi = 0; mi < 2; ++mi)
            for (int ni = 0; ni < 4; ++ni)
                acc[mi][ni] = __builtin_amdgcn_wmma_f32_16x16x32_bf16(
                    false, af[mi].v, false, bfr[ni].v, (short)0, acc[mi][ni],
                    false, false);
        wait_async_tiles();
        __syncthreads();
    }

    // Epilogue.  C/D layout: n = lane&15, m = j + 8*(lane>>4)
    int coln = lane & 15;
    int rhi  = (lane & 16) ? 8 : 0;
    for (int ni = 0; ni < 4; ++ni) {
        int n = n0 + wn * 64 + ni * 16 + coln;
        float bv = (bias != nullptr && n < N) ? bias[n] : 0.0f;
        for (int mi = 0; mi < 2; ++mi) {
            for (int j = 0; j < 8; ++j) {
                int m = m0 + wm * 32 + mi * 16 + rhi + j;
                if (m < M && n < N) {
                    float v = acc[mi][ni][j] + bv;
                    if (ACT == 1) v = gelu_exact(v);
                    long long idx = coff + (long long)m * ldc + n;
                    if (OUT_MODE == 0)      ((u16*)Cv)[idx]   = f2bf(v);
                    else if (OUT_MODE == 1) ((float*)Cv)[idx] = v;
                    else                    ((float*)Cv)[idx] += v;
                }
            }
        }
    }
}

// ---------------------------------------------------------------------------
// Weight transpose + fp32 -> bf16:  dst[z][n][k] = src[z][k][n]
// ---------------------------------------------------------------------------
__global__ void transpose_bf16(const float* __restrict__ src,
                               u16* __restrict__ dst, int K, int N) {
    long long base = (long long)blockIdx.z * K * N;
    long long idx  = (long long)blockIdx.x * 256 + threadIdx.x;
    if (idx >= (long long)K * N) return;
    int k = (int)(idx % K);
    int n = (int)(idx / K);
    dst[base + (long long)n * K + k] = f2bf(src[base + (long long)k * N + n]);
}

// ---------------------------------------------------------------------------
// LayerNorm rows (wave per row).  OUT_F32: 1 -> f32, 0 -> bf16
// ---------------------------------------------------------------------------
template <int OUT_F32>
__global__ __launch_bounds__(256) void ln_rows(const float* __restrict__ x,
                                               const float* __restrict__ g,
                                               const float* __restrict__ b,
                                               void* __restrict__ out, int M, int d) {
    int wave = threadIdx.x >> 5, lane = threadIdx.x & 31;
    long long row = (long long)blockIdx.x * 8 + wave;
    if (row >= M) return;
    const float* xr = x + row * d;
    float s = 0.f, q = 0.f;
    for (int c = lane; c < d; c += 32) { float v = xr[c]; s += v; q += v * v; }
    for (int off = 16; off; off >>= 1) {
        s += __shfl_xor(s, off, 32);
        q += __shfl_xor(q, off, 32);
    }
    float mean = s / d;
    float inv  = rsqrtf(q / d - mean * mean + 1e-5f);
    for (int c = lane; c < d; c += 32) {
        float v = (xr[c] - mean) * inv * g[c] + b[c];
        if (OUT_F32) ((float*)out)[row * d + c] = v;
        else         ((u16*)out)[row * d + c]   = f2bf(v);
    }
}

// ---------------------------------------------------------------------------
// Patch-merge gather + LayerNorm -> bf16  (wave per output row)
//   out row r = (((b*196+p)*io+a)*io+q), channel c: blk=c/d_in, cc=c%d_in,
//   src inner (ii,jj) = (2a+(blk&1), 2q+(blk>>1)) in a (2*io x 2*io) grid.
// ---------------------------------------------------------------------------
__global__ __launch_bounds__(256) void merge_ln(const float* __restrict__ h,
                                                const float* __restrict__ g,
                                                const float* __restrict__ bta,
                                                u16* __restrict__ out,
                                                int M, int NP, int io, int d_in) {
    int wave = threadIdx.x >> 5, lane = threadIdx.x & 31;
    long long row = (long long)blockIdx.x * 8 + wave;
    if (row >= M) return;
    int d_out = d_in * 4;
    long long r2 = row;
    int q = (int)(r2 % io); r2 /= io;
    int a = (int)(r2 % io); r2 /= io;
    int p = (int)(r2 % NP);
    int bi = (int)(r2 / NP);
    int in_inner = 2 * io;
    float vals[32];
    int cnt = d_out / 32;
    float s = 0.f, sq = 0.f;
    for (int j = 0; j < cnt; ++j) {
        int c  = lane + 32 * j;
        int blk = c / d_in, cc = c % d_in;
        int ii = 2 * a + (blk & 1);
        int jj = 2 * q + (blk >> 1);
        long long srow = (((long long)bi * NP + p) * in_inner + ii) * in_inner + jj;
        float v = h[srow * d_in + cc];
        vals[j] = v; s += v; sq += v * v;
    }
    for (int off = 16; off; off >>= 1) {
        s  += __shfl_xor(s, off, 32);
        sq += __shfl_xor(sq, off, 32);
    }
    float mean = s / d_out;
    float inv  = rsqrtf(sq / d_out - mean * mean + 1e-5f);
    for (int j = 0; j < cnt; ++j) {
        int c = lane + 32 * j;
        out[row * d_out + c] = f2bf((vals[j] - mean) * inv * g[c] + bta[c]);
    }
}

// ---------------------------------------------------------------------------
// 4x4/s4 patch conv, output laid out as (b,196,4,4,128) rows, fp32 (pre-LN)
// ---------------------------------------------------------------------------
__global__ __launch_bounds__(256) void patch_conv(const float* __restrict__ x,
                                                  const float* __restrict__ w,
                                                  const float* __restrict__ bias,
                                                  float* __restrict__ out) {
    long long idx = (long long)blockIdx.x * 256 + threadIdx.x;  // r*128 + c
    int c = (int)(idx & 127);
    long long r = idx >> 7;
    int iw = (int)(r & 3), ih = (int)((r >> 2) & 3);
    long long pr = r >> 4;
    int p = (int)(pr % 196);
    int b = (int)(pr / 196);
    int hp = p / 14, wp = p % 14;
    int oh = hp * 4 + ih, ow = wp * 4 + iw;
    float acc = bias[c];
    for (int ci = 0; ci < 3; ++ci) {
        const float* xp = x + (((long long)b * 3 + ci) * 224 + oh * 4) * 224 + ow * 4;
        const float* wp4 = w + ((long long)c * 3 + ci) * 16;
        for (int kh = 0; kh < 4; ++kh)
            for (int kw = 0; kw < 4; ++kw)
                acc += xp[kh * 224 + kw] * wp4[kh * 4 + kw];
    }
    out[idx] = acc;
}

// ---------------------------------------------------------------------------
// QKV repack: (3136,1536) bf16 -> q (b,h,n,64)*scale, k (b,h,n,64),
//             vT (b,h,64, n) with padded stride 208
// ---------------------------------------------------------------------------
__global__ __launch_bounds__(256) void repack_qkv(const u16* __restrict__ qkv,
                                                  u16* __restrict__ q,
                                                  u16* __restrict__ k,
                                                  u16* __restrict__ vt) {
    long long idx = (long long)blockIdx.x * 256 + threadIdx.x;  // b,n,h,d
    int d = (int)(idx & 63);
    int h = (int)((idx >> 6) & 7);
    long long rem = idx >> 9;
    int n = (int)(rem % 196);
    int b = (int)(rem / 196);
    long long base = ((long long)b * 196 + n) * 1536;
    long long zi   = (long long)(b * 8 + h);
    q[(zi * 196 + n) * 64 + d]  = f2bf(bf2f(qkv[base + h * 64 + d]) * 0.125f);
    k[(zi * 196 + n) * 64 + d]  = qkv[base + 512 + h * 64 + d];
    vt[(zi * 64 + d) * 208 + n] = qkv[base + 1024 + h * 64 + d];
}

// ---------------------------------------------------------------------------
// Softmax over 196 cols with relative-position bias (wave per row)
//   S fp32 (z,196,196);  P bf16 (z,196,208);  rpe (729,8) for this layer.
// ---------------------------------------------------------------------------
__global__ __launch_bounds__(256) void softmax_rpe(const float* __restrict__ S,
                                                   const float* __restrict__ rpe,
                                                   u16* __restrict__ P) {
    int wave = threadIdx.x >> 5, lane = threadIdx.x & 31;
    long long row = (long long)blockIdx.x * 8 + wave;  // z*196 + n
    if (row >= 128LL * 196) return;
    int n = (int)(row % 196);
    long long z = row / 196;
    int h = (int)(z & 7);
    int i1 = n / 14, j1 = n % 14;
    const float* sr = S + row * 196;
    float vals[7];
    float mx = -3.0e38f;
    for (int j = 0; j < 7; ++j) {
        int m = lane + 32 * j;
        if (m < 196) {
            int i2 = m / 14, j2 = m % 14;
            int ridx = (i1 - i2 + 13) * 27 + (j1 - j2 + 13);
            float t = sr[m] + rpe[ridx * 8 + h];
            vals[j] = t;
            mx = fmaxf(mx, t);
        } else vals[j] = -3.0e38f;
    }
    for (int off = 16; off; off >>= 1) mx = fmaxf(mx, __shfl_xor(mx, off, 32));
    float sum = 0.f;
    float ex[7];
    for (int j = 0; j < 7; ++j) {
        int m = lane + 32 * j;
        float e = (m < 196) ? __expf(vals[j] - mx) : 0.f;
        ex[j] = e; sum += e;
    }
    for (int off = 16; off; off >>= 1) sum += __shfl_xor(sum, off, 32);
    float inv = 1.0f / sum;
    u16* pr = P + (z * 196 + n) * 208;
    for (int j = 0; j < 7; ++j) {
        int m = lane + 32 * j;
        if (m < 196) pr[m] = f2bf(ex[j] * inv);
    }
}

// ---------------------------------------------------------------------------
// h += pos_embed ; token mean
// ---------------------------------------------------------------------------
__global__ __launch_bounds__(256) void add_pos(float* __restrict__ h,
                                               const float* __restrict__ pos) {
    long long idx = (long long)blockIdx.x * 256 + threadIdx.x;  // (b,p,c)
    int c = (int)(idx & 511);
    int p = (int)((idx >> 9) % 196);
    h[idx] += pos[p * 512 + c];
}

__global__ __launch_bounds__(256) void token_mean(const float* __restrict__ h,
                                                  float* __restrict__ out) {
    int b = blockIdx.x;
    for (int c = threadIdx.x; c < 512; c += 256) {
        float s = 0.f;
        for (int n = 0; n < 196; ++n)
            s += h[(((long long)b * 196) + n) * 512 + c];
        out[b * 512 + c] = s * (1.0f / 196.0f);
    }
}

// ---------------------------------------------------------------------------
// GEMM launcher
// ---------------------------------------------------------------------------
static void launch_gemm(hipStream_t st, int out_mode, int act,
                        const u16* A, const u16* B, const float* bias, void* C,
                        int M, int N, int K, int lda, int ldb, int ldc,
                        int Z = 1, long long a_bs = 0, long long b_bs = 0,
                        int c_div = 1, long long c_bs1 = 0, long long c_bs2 = 0) {
    dim3 grid((N + TN - 1) / TN, (M + TM - 1) / TM, Z), blk(256);
    if (out_mode == 0 && act == 0)
        gemm_bf16<0, 0><<<grid, blk, 0, st>>>(A, B, bias, C, M, N, K, lda, ldb, ldc, a_bs, b_bs, c_div, c_bs1, c_bs2);
    else if (out_mode == 0 && act == 1)
        gemm_bf16<0, 1><<<grid, blk, 0, st>>>(A, B, bias, C, M, N, K, lda, ldb, ldc, a_bs, b_bs, c_div, c_bs1, c_bs2);
    else if (out_mode == 1)
        gemm_bf16<1, 0><<<grid, blk, 0, st>>>(A, B, bias, C, M, N, K, lda, ldb, ldc, a_bs, b_bs, c_div, c_bs1, c_bs2);
    else
        gemm_bf16<2, 0><<<grid, blk, 0, st>>>(A, B, bias, C, M, N, K, lda, ldb, ldc, a_bs, b_bs, c_div, c_bs1, c_bs2);
}

// ---------------------------------------------------------------------------
// Orchestration
// ---------------------------------------------------------------------------
extern "C" void kernel_launch(void* const* d_in, const int* in_sizes, int n_in,
                              void* d_out, int out_size, void* d_ws, size_t ws_size,
                              hipStream_t stream) {
    const float* X      = (const float*)d_in[0];
    const float* pcw    = (const float*)d_in[1];
    const float* pcb    = (const float*)d_in[2];
    const float* pln_g  = (const float*)d_in[3];
    const float* pln_b  = (const float*)d_in[4];
    const float* s1_lng = (const float*)d_in[5];
    const float* s1_lnb = (const float*)d_in[6];
    const float* s1_w1  = (const float*)d_in[7];
    const float* s1_b1  = (const float*)d_in[8];
    const float* s1_w2  = (const float*)d_in[9];
    const float* s1_b2  = (const float*)d_in[10];
    const float* m1_lng = (const float*)d_in[11];
    const float* m1_lnb = (const float*)d_in[12];
    const float* m1_w   = (const float*)d_in[13];
    const float* s2_lng = (const float*)d_in[14];
    const float* s2_lnb = (const float*)d_in[15];
    const float* s2_w1  = (const float*)d_in[16];
    const float* s2_b1  = (const float*)d_in[17];
    const float* s2_w2  = (const float*)d_in[18];
    const float* s2_b2  = (const float*)d_in[19];
    const float* m2_lng = (const float*)d_in[20];
    const float* m2_lnb = (const float*)d_in[21];
    const float* m2_w   = (const float*)d_in[22];
    const float* pos    = (const float*)d_in[23];
    const float* ln1_g  = (const float*)d_in[24];
    const float* ln1_b  = (const float*)d_in[25];
    const float* qkv_w  = (const float*)d_in[26];
    const float* qkv_b  = (const float*)d_in[27];
    const float* rpe    = (const float*)d_in[28];
    const float* proj_w = (const float*)d_in[29];
    const float* proj_b = (const float*)d_in[30];
    const float* ln2_g  = (const float*)d_in[31];
    const float* ln2_b  = (const float*)d_in[32];
    const float* mn_w1  = (const float*)d_in[33];
    const float* mn_b1  = (const float*)d_in[34];
    const float* mn_w2  = (const float*)d_in[35];
    const float* mn_b2  = (const float*)d_in[36];
    const float* fc_g   = (const float*)d_in[37];
    const float* fc_b   = (const float*)d_in[38];
    const float* head_w = (const float*)d_in[39];
    const float* head_b = (const float*)d_in[40];

    char* ws = (char*)d_ws;
    size_t cur = 0;
    auto alloc = [&](size_t bytes) -> void* {
        void* p = ws + cur;
        cur += (bytes + 255) & ~(size_t)255;
        return p;
    };

    // bf16 transposed weights
    u16* s1w1t  = (u16*)alloc((size_t)8 * 384 * 128 * 2);
    u16* s1w2t  = (u16*)alloc((size_t)8 * 128 * 384 * 2);
    u16* m1wt   = (u16*)alloc((size_t)256 * 512 * 2);
    u16* s2w1t  = (u16*)alloc((size_t)8 * 768 * 256 * 2);
    u16* s2w2t  = (u16*)alloc((size_t)8 * 256 * 768 * 2);
    u16* m2wt   = (u16*)alloc((size_t)512 * 1024 * 2);
    u16* qkvwt  = (u16*)alloc((size_t)20 * 1536 * 512 * 2);
    u16* projwt = (u16*)alloc((size_t)20 * 512 * 512 * 2);
    u16* mw1t   = (u16*)alloc((size_t)20 * 2048 * 512 * 2);
    u16* mw2t   = (u16*)alloc((size_t)20 * 512 * 2048 * 2);
    u16* headwt = (u16*)alloc((size_t)1000 * 512 * 2);
    // activations
    float* h0   = (float*)alloc((size_t)50176 * 128 * 4);
    float* h1   = (float*)alloc((size_t)50176 * 128 * 4);
    u16*   Abuf = (u16*)alloc((size_t)50176 * 128 * 2);
    u16*   hid  = (u16*)alloc((size_t)50176 * 384 * 2);
    u16*   qb   = (u16*)alloc((size_t)128 * 196 * 64 * 2);
    u16*   kb   = (u16*)alloc((size_t)128 * 196 * 64 * 2);
    u16*   vtb  = (u16*)alloc((size_t)128 * 64 * 208 * 2);
    float* Sb   = (float*)alloc((size_t)128 * 196 * 196 * 4);
    u16*   Pb   = (u16*)alloc((size_t)128 * 196 * 208 * 2);
    u16*   ob   = (u16*)alloc((size_t)3136 * 512 * 2);
    float* mb   = (float*)alloc((size_t)16 * 512 * 4);
    u16*   hA   = (u16*)alloc((size_t)16 * 512 * 2);

    auto tconv = [&](const float* src, u16* dst, int K, int N, int Z) {
        long long tot = (long long)K * N;
        dim3 g((unsigned)((tot + 255) / 256), 1, Z);
        transpose_bf16<<<g, 256, 0, stream>>>(src, dst, K, N);
    };
    tconv(s1_w1, s1w1t, 128, 384, 8);
    tconv(s1_w2, s1w2t, 384, 128, 8);
    tconv(m1_w,  m1wt,  512, 256, 1);
    tconv(s2_w1, s2w1t, 256, 768, 8);
    tconv(s2_w2, s2w2t, 768, 256, 8);
    tconv(m2_w,  m2wt,  1024, 512, 1);
    tconv(qkv_w, qkvwt, 512, 1536, 20);
    tconv(proj_w, projwt, 512, 512, 20);
    tconv(mn_w1, mw1t, 512, 2048, 20);
    tconv(mn_w2, mw2t, 2048, 512, 20);
    tconv(head_w, headwt, 512, 1000, 1);

    // --- patch embed: conv -> h1 (fp32), then patch LN -> h0 (fp32 residual)
    patch_conv<<<25088, 256, 0, stream>>>(X, pcw, pcb, h1);
    ln_rows<1><<<6272, 256, 0, stream>>>(h1, pln_g, pln_b, h0, 50176, 128);

    // --- stem1: 8 x (h += MLP(LN(h))) at d=128, hidden 384, M=50176
    for (int i = 0; i < 8; ++i) {
        ln_rows<0><<<6272, 256, 0, stream>>>(h0, s1_lng + i * 128, s1_lnb + i * 128,
                                             Abuf, 50176, 128);
        launch_gemm(stream, 0, 1, Abuf, s1w1t + (size_t)i * 384 * 128,
                    s1_b1 + i * 384, hid, 50176, 384, 128, 128, 128, 384);
        launch_gemm(stream, 2, 0, hid, s1w2t + (size_t)i * 128 * 384,
                    s1_b2 + i * 128, h0, 50176, 128, 384, 384, 384, 128);
    }

    // --- merge1: gather+LN -> Abuf (12544 x 512), GEMM -> h1 (12544 x 256)
    merge_ln<<<1568, 256, 0, stream>>>(h0, m1_lng, m1_lnb, Abuf, 12544, 196, 2, 128);
    launch_gemm(stream, 1, 0, Abuf, m1wt, nullptr, h1, 12544, 256, 512, 512, 512, 256);

    // --- stem2: 8 blocks at d=256, hidden 768, M=12544
    for (int i = 0; i < 8; ++i) {
        ln_rows<0><<<1568, 256, 0, stream>>>(h1, s2_lng + i * 256, s2_lnb + i * 256,
                                             Abuf, 12544, 256);
        launch_gemm(stream, 0, 1, Abuf, s2w1t + (size_t)i * 768 * 256,
                    s2_b1 + i * 768, hid, 12544, 768, 256, 256, 256, 768);
        launch_gemm(stream, 2, 0, hid, s2w2t + (size_t)i * 256 * 768,
                    s2_b2 + i * 256, h1, 12544, 256, 768, 768, 768, 256);
    }

    // --- merge2 -> h0 (3136 x 512), + pos embed
    merge_ln<<<392, 256, 0, stream>>>(h1, m2_lng, m2_lnb, Abuf, 3136, 196, 1, 256);
    launch_gemm(stream, 1, 0, Abuf, m2wt, nullptr, h0, 3136, 512, 1024, 1024, 1024, 512);
    add_pos<<<6272, 256, 0, stream>>>(h0, pos);

    // --- 20 main transformer blocks
    for (int l = 0; l < 20; ++l) {
        ln_rows<0><<<392, 256, 0, stream>>>(h0, ln1_g + l * 512, ln1_b + l * 512,
                                            Abuf, 3136, 512);
        launch_gemm(stream, 0, 0, Abuf, qkvwt + (size_t)l * 1536 * 512,
                    qkv_b + l * 1536, hid, 3136, 1536, 512, 512, 512, 1536);
        repack_qkv<<<6272, 256, 0, stream>>>(hid, qb, kb, vtb);
        // scores S = (q*scale) @ k^T : batched over z = b*8+h (128)
        launch_gemm(stream, 1, 0, qb, kb, nullptr, Sb, 196, 196, 64, 64, 64, 196,
                    128, 196 * 64, 196 * 64, 1, 196 * 196, 0);
        softmax_rpe<<<3136, 256, 0, stream>>>(Sb, rpe + (size_t)l * 729 * 8, Pb);
        // O = P @ V : out scattered to (b, n, h*64 + d) with ldc = 512
        launch_gemm(stream, 0, 0, Pb, vtb, nullptr, ob, 196, 64, 196, 208, 208, 512,
                    128, 196 * 208, 64 * 208, 8, 196 * 512, 64);
        launch_gemm(stream, 2, 0, ob, projwt + (size_t)l * 512 * 512,
                    proj_b + l * 512, h0, 3136, 512, 512, 512, 512, 512);
        ln_rows<0><<<392, 256, 0, stream>>>(h0, ln2_g + l * 512, ln2_b + l * 512,
                                            Abuf, 3136, 512);
        launch_gemm(stream, 0, 1, Abuf, mw1t + (size_t)l * 2048 * 512,
                    mn_b1 + l * 2048, hid, 3136, 2048, 512, 512, 512, 2048);
        launch_gemm(stream, 2, 0, hid, mw2t + (size_t)l * 512 * 2048,
                    mn_b2 + l * 512, h0, 3136, 512, 2048, 2048, 2048, 512);
    }

    // --- head: mean over tokens, fc-norm, classifier
    token_mean<<<16, 256, 0, stream>>>(h0, mb);
    ln_rows<0><<<2, 256, 0, stream>>>(mb, fc_g, fc_b, hA, 16, 512);
    launch_gemm(stream, 1, 0, hA, headwt, head_b, d_out, 16, 1000, 512,
                512, 512, 1000);
}
